// S4_36060545417897
// MI455X (gfx1250) — matrix operand order repaired and checked by reading, not compiled, not run on hardware
//
#include <hip/hip_runtime.h>
#include <hip/hip_bf16.h>
#include <math.h>

typedef __attribute__((ext_vector_type(16))) _Float16 v16h;
typedef __attribute__((ext_vector_type(2)))  _Float16 v2h;
typedef __attribute__((ext_vector_type(8)))  float    v8f;

#define BB  4
#define HH  512
#define LL  4096
#define NN  32
#define DMM 512
#define OO  (2 * DMM)   // 1024

// ---------------------------------------------------------------------------
// Kernel 1a: per-(h,n) ZOH discretization.  dA = exp(dt*A),
// dB = (dA-1)/A * B,  w = 2*C*dB  (complex, conjugate-pair factor 2).
// ---------------------------------------------------------------------------
__global__ __launch_bounds__(256) void s4_prep_modes(
    const float* __restrict__ log_dt,
    const float* __restrict__ A_re, const float* __restrict__ A_im,
    const float* __restrict__ B_re, const float* __restrict__ B_im,
    const float* __restrict__ C_re, const float* __restrict__ C_im,
    float* __restrict__ dAre, float* __restrict__ dAim,
    float* __restrict__ wre,  float* __restrict__ wim) {
  int idx = blockIdx.x * blockDim.x + threadIdx.x;
  if (idx >= HH * NN) return;
  int h = idx / NN;
  float dt = expf(log_dt[h]);
  float ar = A_re[idx], ai = A_im[idx];
  float tr = dt * ar, ti = dt * ai;
  float mg = expf(tr);
  float er = mg * cosf(ti), ei = mg * sinf(ti);          // dA
  float nr = er - 1.0f, ni = ei;                         // dA - 1
  float inv = 1.0f / (ar * ar + ai * ai);
  float dbr = (nr * ar + ni * ai) * inv;                 // (dA-1)/A
  float dbi = (ni * ar - nr * ai) * inv;
  float br = B_re[idx], bi = B_im[idx];
  float dBr = dbr * br - dbi * bi;                       // * B
  float dBi = dbr * bi + dbi * br;
  float cr = C_re[idx], ci = C_im[idx];                  // C channel 0
  wre[idx]  = 2.0f * (cr * dBr - ci * dBi);
  wim[idx]  = 2.0f * (cr * dBi + ci * dBr);
  dAre[idx] = er;
  dAim[idx] = ei;
}

// ---------------------------------------------------------------------------
// Kernel 1b: W_out fp32 -> f16 copy (B-operand of the WMMA GEMM).
// ---------------------------------------------------------------------------
__global__ __launch_bounds__(256) void s4_prep_w(
    const float* __restrict__ W, _Float16* __restrict__ Wh) {
  int idx = blockIdx.x * blockDim.x + threadIdx.x;
  if (idx < OO * HH) Wh[idx] = (_Float16)W[idx];
}

// ---------------------------------------------------------------------------
// Kernel 2: diagonal SSM scan (exactly equals the zero-padded FFT conv),
// + D skip + exact GELU, emitting f16 activations.
// One wave32 per (b,h) sequence, one lane per mode.  Per 32-step chunk the
// per-mode partials are transposed through LDS and row-reduced, avoiding a
// cross-lane butterfly every step.
// ---------------------------------------------------------------------------
__global__ __launch_bounds__(256) void s4_scan(
    const float* __restrict__ u,
    const float* __restrict__ dAre, const float* __restrict__ dAim,
    const float* __restrict__ wre,  const float* __restrict__ wim,
    const float* __restrict__ D,
    _Float16* __restrict__ yact) {
  __shared__ float red[8][32][33];      // [wave][step-in-chunk][mode], padded
  int tid  = threadIdx.x;
  int wv   = tid >> 5;
  int lane = tid & 31;
  int seq  = blockIdx.x * 8 + wv;       // (b,h) flat
  int b = seq / HH, h = seq % HH;
  int mi = h * NN + lane;
  float ar = dAre[mi], ai = dAim[mi];
  float cwr = wre[mi], cwi = wim[mi];
  float Dh = D[h];
  const float* up = u + (size_t)(b * HH + h) * LL;
  _Float16*    yp = yact + (size_t)(b * HH + h) * LL;
  float sr = 0.0f, si = 0.0f;
  for (int l0 = 0; l0 < LL; l0 += 32) {
    float uc = up[l0 + lane];                       // coalesced chunk load
#pragma unroll
    for (int j = 0; j < 32; ++j) {
      float uj  = __shfl(uc, j, 32);                // v_readlane (const idx)
      float pre = sr;
      sr = fmaf(ar, pre, fmaf(-ai, si, uj));        // s = dA*s + u
      si = fmaf(ar, si, ai * pre);
      red[wv][j][lane] = fmaf(cwr, sr, -cwi * si);  // Re(w * s)
    }
    __syncthreads();
    float acc = 0.0f;
#pragma unroll
    for (int k = 0; k < 32; ++k) acc += red[wv][lane][k];
    float y = acc + Dh * uc;                        // lane's l = l0+lane
    float g = 0.5f * y * (1.0f + erff(y * 0.70710678118654752f));
    yp[l0 + lane] = (_Float16)g;
    __syncthreads();
  }
}

// ---------------------------------------------------------------------------
// Kernel 3: output projection via v_wmma_f32_16x16x32_f16 + bias + GLU.
// GEMM: D[m][n] = sum_k A[m=pos][k=h] * B[k][n],  B[k][n] = W_out[n][k].
// A tile (16 x 512 f16) staged into LDS in exact A-fragment order so each
// lane reads its 32B fragment with two ds_load_b128.  B fragments read
// directly (L2-resident): lane gets 16 contiguous K at fixed n.
// Each wave owns channel pair (n0, n0+512) -> fused GLU, then an LDS
// transpose yields coalesced (b, o, l) stores.
// ---------------------------------------------------------------------------
__global__ __launch_bounds__(256) void s4_gemm(
    const _Float16* __restrict__ yact,
    const _Float16* __restrict__ Wh,
    const float* __restrict__ b_out,
    float* __restrict__ out) {
  __shared__ __align__(32) _Float16 ldsA[16 * 512];  // 16KB, frag-swizzled
  __shared__ float ldsO[8 * 256];                    // 8KB, epilogue transpose
  int tid  = threadIdx.x;
  int wv   = tid >> 5;
  int lane = tid & 31;
  int mg = blockIdx.x * 16;          // flat position tile (b*L + l)
  int b  = mg / LL;
  int l0 = mg % LL;
  const _Float16* ybase = yact + (size_t)(b * HH) * LL + l0;

  // ---- phase 1: cooperative A-tile load with fragment swizzle ----
  // element (k, m):  t=k/32; kk=k%32; hi=kk>>4; j=kk&15
  // -> halfword index  t*512 + (m + 16*hi)*16 + j
#pragma unroll
  for (int it = 0; it < 16; ++it) {
    int i = tid + 256 * it;          // 0..4095 half-pairs
    int k = i >> 3;
    int m = (i & 7) * 2;
    v2h v = *(const v2h*)(ybase + (size_t)k * LL + m);
    int t = k >> 5, kk = k & 31;
    int hi = kk >> 4, j = kk & 15;
    int base = t * 512 + (m + 16 * hi) * 16 + j;
    ldsA[base]      = v[0];
    ldsA[base + 16] = v[1];
  }
  __syncthreads();

  // ---- phase 2: K-loop of WMMAs ----
  int n0 = blockIdx.y * 128 + wv * 16;   // n0 in [0, 512)
  int nl = lane & 15, lh = lane >> 4;
  const _Float16* wb0 = Wh + (size_t)(n0 + nl) * HH + 16 * lh;
  const _Float16* wb1 = Wh + (size_t)(n0 + 512 + nl) * HH + 16 * lh;
  v8f c0 = {}, c1 = {};
#pragma unroll
  for (int t = 0; t < 16; ++t) {
    v16h a   = *(const v16h*)(&ldsA[t * 512 + lane * 16]);
    v16h bf0 = *(const v16h*)(wb0 + t * 32);
    v16h bf1 = *(const v16h*)(wb1 + t * 32);
    c0 = __builtin_amdgcn_wmma_f32_16x16x32_f16(false, a, false, bf0,
                                                (short)0, c0, false, false);
    c1 = __builtin_amdgcn_wmma_f32_16x16x32_f16(false, a, false, bf1,
                                                (short)0, c1, false, false);
  }

  // ---- epilogue: bias + GLU, LDS transpose, coalesced store ----
  float bo0 = b_out[n0 + nl];
  float bo1 = b_out[n0 + 512 + nl];
  float* po = ldsO + wv * 256;
#pragma unroll
  for (int r = 0; r < 8; ++r) {
    int ml = r + 8 * lh;                       // C/D layout: M = r (+8 hi half)
    float z0 = c0[r] + bo0;
    float z1 = c1[r] + bo1;
    po[nl * 16 + ml] = z0 * (1.0f / (1.0f + expf(-z1)));   // GLU
  }
  __syncthreads();
  float* ob = out + (size_t)(b * DMM) * LL + l0;
#pragma unroll
  for (int jj = 0; jj < 8; ++jj) {
    int row = jj * 2 + lh;                     // output channel n0+row
    ob[(size_t)(n0 + row) * LL + nl] = po[row * 16 + nl];
  }
}

// ---------------------------------------------------------------------------
extern "C" void kernel_launch(void* const* d_in, const int* in_sizes, int n_in,
                              void* d_out, int out_size, void* d_ws, size_t ws_size,
                              hipStream_t stream) {
  (void)in_sizes; (void)n_in; (void)out_size; (void)ws_size;
  const float* u      = (const float*)d_in[0];
  const float* log_dt = (const float*)d_in[1];
  const float* A_re   = (const float*)d_in[2];
  const float* A_im   = (const float*)d_in[3];
  const float* B_re   = (const float*)d_in[4];
  const float* B_im   = (const float*)d_in[5];
  const float* C_re   = (const float*)d_in[6];
  const float* C_im   = (const float*)d_in[7];
  const float* D      = (const float*)d_in[8];
  const float* W_out  = (const float*)d_in[9];
  const float* b_out  = (const float*)d_in[10];
  float* out = (float*)d_out;

  // workspace layout (all offsets 32B aligned):
  float* dAre = (float*)d_ws;                 // 16384 f32
  float* dAim = dAre + HH * NN;
  float* wre  = dAim + HH * NN;
  float* wim  = wre  + HH * NN;               // ends at 256KB
  _Float16* Wh   = (_Float16*)(wim + HH * NN);        // 512K f16 (1MB)
  _Float16* yact = Wh + (size_t)OO * HH;              // 8.4M f16 (16MB)

  s4_prep_modes<<<(HH * NN + 255) / 256, 256, 0, stream>>>(
      log_dt, A_re, A_im, B_re, B_im, C_re, C_im, dAre, dAim, wre, wim);
  s4_prep_w<<<(OO * HH + 255) / 256, 256, 0, stream>>>(W_out, Wh);
  s4_scan<<<(BB * HH) / 8, 256, 0, stream>>>(u, dAre, dAim, wre, wim, D, yact);
  s4_gemm<<<dim3((BB * LL) / 16, 4), 256, 0, stream>>>(yact, Wh, b_out, out);
}
